// MultiAgentAttention_11862699672176
// MI455X (gfx1250) — matrix operand order repaired and checked by reading, not compiled, not run on hardware
//
#include <hip/hip_runtime.h>

// MI455X / gfx1250 fused multi-head attention block.
// B=32, N=512, E=256, H=8, HD=32.
// GEMMs: v_wmma_f32_16x16x32_f16.  K/V staging: tensor_load_to_lds (TDM)
// with hardware LDS row padding; completion via s_wait_tensorcnt.
// Workspace (48 MB):
//   [0,        8MB)  q   f16 [B,H,N,HD]  (pre-scaled by HD^-0.5)
//   [8MB,     16MB)  k   f16 [B,H,N,HD]
//   [16MB,    24MB)  vT  f16 [B,H,HD,N]  (transposed for PV B-fragments)
//   [24MB,    32MB)  o   f16 [B,N,E]
//   [32MB,    48MB)  y   f32 [B,N,E]

typedef __attribute__((ext_vector_type(16))) _Float16 v16h;
typedef __attribute__((ext_vector_type(8)))  _Float16 v8h;
typedef __attribute__((ext_vector_type(8)))  float    v8f;
typedef __attribute__((ext_vector_type(4)))  unsigned int v4u;
typedef __attribute__((ext_vector_type(8)))  int v8i_t;
typedef __attribute__((ext_vector_type(4)))  int v4i_t;

__device__ __forceinline__ v8f wmma_f32(v16h a, v16h b, v8f c) {
  return __builtin_amdgcn_wmma_f32_16x16x32_f16(false, a, false, b, (short)0, c,
                                                false, false);
}

__device__ __forceinline__ v16h cat16(v8h lo, v8h hi) {
  return __builtin_shufflevector(lo, hi, 0, 1, 2, 3, 4, 5, 6, 7, 8, 9, 10, 11,
                                 12, 13, 14, 15);
}

// A fragment (16x32 f16) from row-major [m][k] LDS, leading dim ld (mult of 8).
// Lane l: M = l&15, half = l>>4.  elem i -> K = (i>>3)*16 + half*8 + (i&7).
// Two 16B ds_load_b128 per lane.
__device__ __forceinline__ v16h frag_a(const _Float16* p, int row0, int k0,
                                       int ld, int lane) {
  const _Float16* rp =
      p + (row0 + (lane & 15)) * ld + k0 + ((lane >> 4) << 3);
  v8h lo = *(const v8h*)rp;         // K = half*8 .. +7
  v8h hi = *(const v8h*)(rp + 16);  // K = 16 + half*8 .. +7
  return cat16(lo, hi);
}

// B fragment (32x16 f16) where storage is [n][k] row-major, leading dim ld.
// Lane l: N = n0 + (l&15), half = l>>4.  elem i -> K = (i>>3)*16 + half*8 + (i&7).
__device__ __forceinline__ v16h frag_bt(const _Float16* p, int n0, int k0,
                                        int ld, int lane) {
  const _Float16* rp =
      p + (n0 + (lane & 15)) * ld + k0 + ((lane >> 4) << 3);
  v8h lo = *(const v8h*)rp;
  v8h hi = *(const v8h*)(rp + 16);
  return cat16(lo, hi);
}

// ---------------------------------------------------------------------------
// Tensor Data Mover: 2D tile load Global -> LDS, optional per-row LDS padding.
// Dims/strides in data_size units (data_size code 3 = 8 bytes).
// ---------------------------------------------------------------------------
__device__ __forceinline__ void tdm_load_2d(unsigned lds_off, const void* gptr,
                                            unsigned tensor_d0,
                                            unsigned long long d0_stride,
                                            unsigned tensor_d1, unsigned tile_d0,
                                            unsigned tile_d1, unsigned ds_code,
                                            unsigned pad_interval,
                                            unsigned pad_amount, bool pad_en) {
  const unsigned long long ga = (unsigned long long)gptr;
  v4u g0;
  g0[0] = 1u;  // count=1, user descriptor
  g0[1] = lds_off;
  g0[2] = (unsigned)(ga & 0xFFFFFFFFu);
  g0[3] = (unsigned)((ga >> 32) & 0x1FFFFFFu) | 0x80000000u;  // type=2 (image)

  unsigned long long q0 = ((unsigned long long)ds_code << 16) |
                          ((unsigned long long)(pad_en ? 1u : 0u) << 20) |
                          ((unsigned long long)pad_interval << 22) |
                          ((unsigned long long)pad_amount << 25) |
                          ((unsigned long long)(tensor_d0 & 0xFFFFu) << 48);
  unsigned long long q1 = ((unsigned long long)tensor_d0 >> 16) |
                          ((unsigned long long)tensor_d1 << 16) |
                          ((unsigned long long)tile_d0 << 48);
  unsigned long long q2 = (unsigned long long)tile_d1 |
                          ((d0_stride & 0xFFFFFFFFull) << 32);
  unsigned long long q3 = (d0_stride >> 32);  // tensor_dim1_stride = 0 (2D)
  union {
    unsigned long long q[4];
    v8i_t v;
  } g1;
  g1.q[0] = q0; g1.q[1] = q1; g1.q[2] = q2; g1.q[3] = q3;
  v4i_t z4 = {0, 0, 0, 0};
#if __clang_major__ >= 23
  v8i_t z8 = {};
  __builtin_amdgcn_tensor_load_to_lds(g0, g1.v, z4, z4, z8, 0);
#else
  __builtin_amdgcn_tensor_load_to_lds(g0, g1.v, z4, z4, 0);
#endif
}

// ---------------------------------------------------------------------------
// Kernel 1: qkv = x @ in_proj_w^T + in_proj_b -> q,k f16 [B,H,N,HD], vT.
// M=16384, Nout=768, K=256.  Grid (256,12), 64x64 tiles, 8 waves.
// ---------------------------------------------------------------------------
__global__ __launch_bounds__(256) void qkv_proj_kernel(
    const float* __restrict__ x, const float* __restrict__ w,
    const float* __restrict__ bias, _Float16* __restrict__ qh,
    _Float16* __restrict__ kh, _Float16* __restrict__ vt) {
  __shared__ _Float16 As[64 * 40];  // [64 m][40], 32 k used
  __shared__ _Float16 Ws[64 * 40];  // [64 n][40], natural W layout (n-major)
  const int tid = threadIdx.x;
  const int lane = tid & 31;
  const int wv = tid >> 5;
  const int m0 = blockIdx.x * 64;
  const int n0 = blockIdx.y * 64;
  const int rb = wv & 3;   // 4 row blocks of 16
  const int cb = wv >> 2;  // 2 col blocks of 32
  v8f acc0 = {};
  v8f acc1 = {};

  const int r = tid >> 2;
  const int c = (tid & 3) * 8;
  for (int kk = 0; kk < 256; kk += 32) {
    const float* xs = x + (long)(m0 + r) * 256 + kk + c;
    const float* ww = w + (long)(n0 + r) * 256 + kk + c;
    if (kk + 32 < 256) {
      __builtin_prefetch(xs + 32, 0, 3);
      __builtin_prefetch(ww + 32, 0, 3);
    }
    v8h va, vb;
#pragma unroll
    for (int j = 0; j < 8; ++j) va[j] = (_Float16)xs[j];
#pragma unroll
    for (int j = 0; j < 8; ++j) vb[j] = (_Float16)ww[j];
    *(v8h*)(As + r * 40 + c) = va;
    *(v8h*)(Ws + r * 40 + c) = vb;
    __syncthreads();
    v16h a = frag_a(As, rb * 16, 0, 40, lane);
    v16h b0 = frag_bt(Ws, cb * 32, 0, 40, lane);
    v16h b1 = frag_bt(Ws, cb * 32 + 16, 0, 40, lane);
    acc0 = wmma_f32(a, b0, acc0);
    acc1 = wmma_f32(a, b1, acc1);
    __syncthreads();
  }

  const float scale = 0.17677669529663687f;  // HD^-0.5
#pragma unroll
  for (int t = 0; t < 2; ++t) {
    v8f acc = t ? acc1 : acc0;
    const int n = n0 + cb * 32 + t * 16 + (lane & 15);
    const float bv = bias[n];
    const int sec = n >> 8;  // 0=q 1=k 2=v
    const int e = n & 255;
    const int hh = e >> 5, d = e & 31;
#pragma unroll
    for (int i = 0; i < 8; ++i) {
      const int m = m0 + rb * 16 + ((lane >> 4) << 3) + i;
      const int b = m >> 9, nr = m & 511;
      const float val = acc[i] + bv;
      if (sec == 0)
        qh[(((long)(b * 8 + hh) * 512) + nr) * 32 + d] =
            (_Float16)(val * scale);
      else if (sec == 1)
        kh[(((long)(b * 8 + hh) * 512) + nr) * 32 + d] = (_Float16)val;
      else  // v stored transposed: [B,H,HD,N]
        vt[(((long)(b * 8 + hh) * 32) + d) * 512 + nr] = (_Float16)val;
    }
  }
}

// ---------------------------------------------------------------------------
// Kernel 2: attention.  One WG per (b, 32-row q tile); loops 8 heads, K/V
// staged by TDM into padded LDS, head-mean attn weights accumulate in LDS.
// Dynamic LDS ~240 KB of the 320 KB WGP pool.
// ---------------------------------------------------------------------------
__global__ __launch_bounds__(256) void attention_kernel(
    const _Float16* __restrict__ qh, const _Float16* __restrict__ kh,
    const _Float16* __restrict__ vt, const unsigned char* __restrict__ pre_mask,
    _Float16* __restrict__ oh, float* __restrict__ attn_out) {
  extern __shared__ char smem[];
  _Float16* Ks = (_Float16*)smem;             // [512][40]  40 KB (32 used)
  _Float16* Vt = Ks + 512 * 40;               // [32][520]  32.5 KB (512 used)
  _Float16* Qs = Vt + 32 * 520;               // [32][40]    2.5 KB
  float*    Sc = (float*)(Qs + 32 * 40);      // [32][512]  64 KB
  _Float16* Pr = (_Float16*)(Sc + 32 * 512);  // [32][520]  32.5 KB
  float*    Wa = (float*)(Pr + 32 * 520);     // [32][512]  64 KB
  float*    Op = Wa + 32 * 512;               // [32][32]    4 KB

  const int tid = threadIdx.x, lane = tid & 31, wv = tid >> 5;
  const int b = blockIdx.x >> 4;
  const int q0 = (blockIdx.x & 15) * 32;

  for (int i = tid; i < 32 * 512; i += 256) Wa[i] = 0.f;

#pragma unroll 1
  for (int h = 0; h < 8; ++h) {
    const long base = ((long)(b * 8 + h)) * 512 * 32;
    // TDM: stage K [512 x 64B rows -> 80B padded] and V^T [32 x 1KB rows ->
    // 1040B padded] into LDS; single wave issues, tensorcnt gates the rest.
    if (tid < 32) {
      tdm_load_2d((unsigned)(unsigned long long)(uintptr_t)Ks, kh + base,
                  /*t_d0=*/8, /*stride=*/8, /*t_d1=*/512, /*tile_d0=*/8,
                  /*tile_d1=*/512, /*8B*/ 3, /*pad every 16 dw*/ 3,
                  /*pad 4 dw*/ 3, true);
      tdm_load_2d((unsigned)(unsigned long long)(uintptr_t)Vt, vt + base,
                  /*t_d0=*/128, /*stride=*/128, /*t_d1=*/32, /*tile_d0=*/128,
                  /*tile_d1=*/32, /*8B*/ 3, /*pad every 256 dw*/ 7,
                  /*pad 4 dw*/ 3, true);
      __builtin_amdgcn_s_wait_tensorcnt(0);
    }
    if (tid < 128) {  // Q tile: 32 rows x 32 halves, one v8h per thread
      const int r = tid >> 2, cc = (tid & 3) * 8;
      *(v8h*)(Qs + r * 40 + cc) =
          *(const v8h*)(qh + base + (long)(q0 + r) * 32 + cc);
    }
    __syncthreads();

    // scores = Q K^T : wave -> (rb = wv&1, 8 key blocks of 16)
    {
      const int rb = wv & 1;
      v16h a = frag_a(Qs, rb * 16, 0, 40, lane);
#pragma unroll 1
      for (int j = 0; j < 8; ++j) {
        const int n0 = ((wv >> 1) * 8 + j) * 16;
        v16h bf = frag_bt(Ks, n0, 0, 40, lane);  // B[d][key] from K[key][d]
        v8f cacc = {};
        cacc = wmma_f32(a, bf, cacc);
        const int key = n0 + (lane & 15);
#pragma unroll
        for (int i = 0; i < 8; ++i)
          Sc[(rb * 16 + ((lane >> 4) << 3) + i) * 512 + key] = cacc[i];
      }
    }
    __syncthreads();

    // mask + softmax: 4 rows per wave, shuffle reductions (wave32)
    {
      const unsigned char* mrow =
          pre_mask + ((long)(b * 8 + h) * 512 + q0) * 512;
#pragma unroll 1
      for (int rr = 0; rr < 4; ++rr) {
        const int r = wv * 4 + rr;
        float* row = Sc + r * 512;
        const unsigned char* mr = mrow + (long)r * 512;
        float mx = -1e30f;
        for (int ci = lane; ci < 512; ci += 32) {
          const float val = mr[ci] ? -1e9f : row[ci];
          row[ci] = val;  // write masked score back (this lane owns ci)
          mx = fmaxf(mx, val);
        }
#pragma unroll
        for (int off = 16; off > 0; off >>= 1)
          mx = fmaxf(mx, __shfl_xor(mx, off, 32));
        float s = 0.f;
        for (int ci = lane; ci < 512; ci += 32) s += __expf(row[ci] - mx);
#pragma unroll
        for (int off = 16; off > 0; off >>= 1) s += __shfl_xor(s, off, 32);
        const float inv = 1.0f / s;
        for (int ci = lane; ci < 512; ci += 32) {
          const float p = __expf(row[ci] - mx) * inv;
          Pr[r * 520 + ci] = (_Float16)p;
          Wa[r * 512 + ci] += p;  // head-mean accumulation
        }
      }
    }
    __syncthreads();

    // o = P @ V : split-K across waves, 2x2 grid of 16x16 tiles
    {
      const int rb = wv & 1, cb = (wv >> 1) & 1, kh2 = wv >> 2;
      v8f cacc = {};
#pragma unroll 1
      for (int t = 0; t < 8; ++t) {
        const int k0 = kh2 * 256 + t * 32;
        v16h a = frag_a(Pr, rb * 16, k0, 520, lane);
        v16h bf = frag_bt(Vt, cb * 16, k0, 520, lane);  // B[key][d] from V^T
        cacc = wmma_f32(a, bf, cacc);
      }
      if (kh2 == 1) {
#pragma unroll
        for (int i = 0; i < 8; ++i)
          Op[(rb * 16 + ((lane >> 4) << 3) + i) * 32 + cb * 16 + (lane & 15)] =
              cacc[i];
      }
      __syncthreads();
      if (kh2 == 0) {
#pragma unroll
        for (int i = 0; i < 8; ++i) {
          const int r = rb * 16 + ((lane >> 4) << 3) + i;
          const int d = cb * 16 + (lane & 15);
          const float v = cacc[i] + Op[r * 32 + d];
          oh[((long)(b * 512 + q0 + r)) * 256 + h * 32 + d] = (_Float16)v;
        }
      }
    }
    __syncthreads();
  }

  for (int i = tid; i < 32 * 512; i += 256) {
    const int r = i >> 9, ci = i & 511;
    attn_out[((long)(b * 512 + q0 + r)) * 512 + ci] = Wa[i] * 0.125f;
  }
}

// ---------------------------------------------------------------------------
// Kernel 3: y = o @ out_w^T + out_b   (M=16384, N=256, K=256)
// ---------------------------------------------------------------------------
__global__ __launch_bounds__(256) void out_proj_kernel(
    const _Float16* __restrict__ oh, const float* __restrict__ w,
    const float* __restrict__ bias, float* __restrict__ y) {
  __shared__ _Float16 As[64 * 40];
  __shared__ _Float16 Ws[64 * 40];
  const int tid = threadIdx.x;
  const int lane = tid & 31;
  const int wv = tid >> 5;
  const int m0 = blockIdx.x * 64;
  const int n0 = blockIdx.y * 64;
  const int rb = wv & 3;
  const int cb = wv >> 2;
  v8f acc0 = {};
  v8f acc1 = {};

  const int r = tid >> 2;
  const int c = (tid & 3) * 8;
  for (int kk = 0; kk < 256; kk += 32) {
    const _Float16* osrc = oh + (long)(m0 + r) * 256 + kk + c;
    const float* ww = w + (long)(n0 + r) * 256 + kk + c;
    if (kk + 32 < 256) {
      __builtin_prefetch(osrc + 32, 0, 3);
      __builtin_prefetch(ww + 32, 0, 3);
    }
    *(v8h*)(As + r * 40 + c) = *(const v8h*)osrc;
    v8h vb;
#pragma unroll
    for (int j = 0; j < 8; ++j) vb[j] = (_Float16)ww[j];
    *(v8h*)(Ws + r * 40 + c) = vb;
    __syncthreads();
    v16h a = frag_a(As, rb * 16, 0, 40, lane);
    v16h b0 = frag_bt(Ws, cb * 32, 0, 40, lane);
    v16h b1 = frag_bt(Ws, cb * 32 + 16, 0, 40, lane);
    acc0 = wmma_f32(a, b0, acc0);
    acc1 = wmma_f32(a, b1, acc1);
    __syncthreads();
  }

#pragma unroll
  for (int t = 0; t < 2; ++t) {
    v8f acc = t ? acc1 : acc0;
    const int n = n0 + cb * 32 + t * 16 + (lane & 15);
    const float bv = bias[n];
#pragma unroll
    for (int i = 0; i < 8; ++i) {
      const int m = m0 + rb * 16 + ((lane >> 4) << 3) + i;
      y[(long)m * 256 + n] = acc[i] + bv;
    }
  }
}

// ---------------------------------------------------------------------------
// Kernel 4: post mask + residual + LayerNorm.  One wave per row of 256.
// ---------------------------------------------------------------------------
__global__ __launch_bounds__(256) void ln_kernel(
    const float* __restrict__ x, const float* __restrict__ y,
    const unsigned char* __restrict__ pmask, const float* __restrict__ gamma,
    const float* __restrict__ beta, float* __restrict__ out) {
  const int row = blockIdx.x * 8 + (threadIdx.x >> 5);
  const int lane = threadIdx.x & 31;
  const int b = row >> 9, nr = row & 511;
  const bool masked = pmask[b * 512 + nr] != 0;
  const float* xr = x + (long)row * 256;
  const float* yr = y + (long)row * 256;
  float v[8];
  float s = 0.f;
#pragma unroll
  for (int j = 0; j < 8; ++j) {
    const int ci = lane + j * 32;
    const float t = xr[ci] + (masked ? 0.f : yr[ci]);
    v[j] = t;
    s += t;
  }
#pragma unroll
  for (int off = 16; off > 0; off >>= 1) s += __shfl_xor(s, off, 32);
  const float mu = s * (1.0f / 256.0f);
  float sq = 0.f;
#pragma unroll
  for (int j = 0; j < 8; ++j) {
    const float d = v[j] - mu;
    sq += d * d;
  }
#pragma unroll
  for (int off = 16; off > 0; off >>= 1) sq += __shfl_xor(sq, off, 32);
  const float inv = rsqrtf(sq * (1.0f / 256.0f) + 1e-5f);
#pragma unroll
  for (int j = 0; j < 8; ++j) {
    const int ci = lane + j * 32;
    out[(long)row * 256 + ci] = (v[j] - mu) * inv * gamma[ci] + beta[ci];
  }
}

// ---------------------------------------------------------------------------
extern "C" void kernel_launch(void* const* d_in, const int* in_sizes, int n_in,
                              void* d_out, int out_size, void* d_ws,
                              size_t ws_size, hipStream_t stream) {
  (void)in_sizes; (void)n_in; (void)out_size; (void)ws_size;
  const float* x = (const float*)d_in[0];
  const unsigned char* pre_mask = (const unsigned char*)d_in[1];   // bool
  const unsigned char* post_mask = (const unsigned char*)d_in[2];  // bool
  const float* in_w = (const float*)d_in[3];
  const float* in_b = (const float*)d_in[4];
  const float* out_w = (const float*)d_in[5];
  const float* out_b = (const float*)d_in[6];
  const float* gamma = (const float*)d_in[7];
  const float* beta = (const float*)d_in[8];

  _Float16* ws16 = (_Float16*)d_ws;
  _Float16* qh = ws16;
  _Float16* kh = ws16 + 4194304;
  _Float16* vt = ws16 + 2 * 4194304;  // [B,H,HD,N]
  _Float16* oh = ws16 + 3 * 4194304;
  float* y = (float*)((char*)d_ws + 33554432);

  float* out = (float*)d_out;           // [B,N,E] f32
  float* attn = out + 32L * 512 * 256;  // [B,N,N] f32

  qkv_proj_kernel<<<dim3(256, 12), 256, 0, stream>>>(x, in_w, in_b, qh, kh, vt);

  const size_t smem2 =
      (512 * 40 + 32 * 520 + 32 * 40 + 32 * 520) * sizeof(_Float16) +
      (32 * 512 + 32 * 512 + 32 * 32) * sizeof(float);
  attention_kernel<<<dim3(512), 256, smem2, stream>>>(qh, kh, vt, pre_mask, oh,
                                                      attn);

  out_proj_kernel<<<dim3(256, 4), 256, 0, stream>>>(oh, out_w, out_b, y);

  ln_kernel<<<dim3(2048), 256, 0, stream>>>(x, y, post_mask, gamma, beta, out);
}
